// EmbeddingLSTMCell_55044300866146
// MI455X (gfx1250) — matrix operand order repaired and checked by reading, not compiled, not run on hardware
//
#include <hip/hip_runtime.h>

#define B_ROWS 8192
#define IN_DIM 1024
#define K_DIM  2048
#define H_DIM  1024
#define KT     32
#define MT     64
#define NT     64
#define NKT    (K_DIM / KT)   // 64 k-tiles
#define NRB    (B_ROWS / MT)  // 128 row blocks
#define NHB    (H_DIM / NT)   // 16 h blocks

// ---- tile layouts (units = 16-bit halves unless noted) ----
#define ASTR   40
#define BSTR   40
#define A_TILE_HALVES 5120
#define A_TILE_BYTES  10240
#define W_TILE_HALVES 20480
#define W_TILE_BYTES  40960
#define STAGE_HALVES  25600
#define STAGE_BYTES   51200
#define A_HI_O 0
#define A_LO_O 2560
#define B_HI_O 5120
#define B_LO_O 15360

#define WS_A_BYTES ((size_t)NRB * NKT * A_TILE_BYTES)            // 80 MB
#define WS_W_BYTES ((size_t)NKT * NHB * W_TILE_BYTES)            // 40 MB
#define WS_NEEDED  (WS_A_BYTES + WS_W_BYTES)                     // 120 MB

typedef __attribute__((ext_vector_type(16))) __bf16 v16bf;
typedef __attribute__((ext_vector_type(8)))  float  v8f;
typedef __attribute__((ext_vector_type(4)))  unsigned int u32x4;
typedef __attribute__((ext_vector_type(8)))  unsigned int u32x8;

struct Frag2 { u32x4 a, b; };

static __device__ __forceinline__ unsigned short f2bf(float f) {
    unsigned int u = __builtin_bit_cast(unsigned int, f);
    u += 0x7FFFu + ((u >> 16) & 1u);   // round-to-nearest-even
    return (unsigned short)(u >> 16);
}
static __device__ __forceinline__ float bf2f(unsigned short h) {
    unsigned int u = ((unsigned int)h) << 16;
    return __builtin_bit_cast(float, u);
}
static __device__ __forceinline__ void split2(float a, float b,
                                              unsigned int& hi, unsigned int& lo) {
    unsigned short ah = f2bf(a), bh = f2bf(b);
    float ar = a - bf2f(ah);
    float br = b - bf2f(bh);
    hi = (unsigned int)ah | ((unsigned int)bh << 16);
    lo = (unsigned int)f2bf(ar) | ((unsigned int)f2bf(br) << 16);
}
static __device__ __forceinline__ v16bf load_frag(const unsigned short* p0,
                                                  const unsigned short* p1) {
    Frag2 fr;
    fr.a = *(const u32x4*)p0;
    fr.b = *(const u32x4*)p1;
    return __builtin_bit_cast(v16bf, fr);
}
static __device__ __forceinline__ float fast_sigmoid(float x) {
    return 1.0f / (1.0f + __expf(-x));
}
static __device__ __forceinline__ float fast_tanh(float x) {
    x = fminf(fmaxf(x, -15.0f), 15.0f);
    float e = __expf(2.0f * x);
    return (e - 1.0f) / (e + 1.0f);
}

// =====================================================================
// Prep pass 1: convert x||h_prev rows into bf16 hi/lo A-tiles in ws
// =====================================================================
__global__ __launch_bounds__(256)
void prep_A(const float* __restrict__ x, const float* __restrict__ h_prev,
            unsigned short* __restrict__ wsA) {
    const int tid = threadIdx.x;
    const int rb  = blockIdx.x;
    const int kt  = blockIdx.y;
    const int ktBase = kt * KT;
    const float* Asrc = (ktBase < IN_DIM) ? (x + ktBase) : (h_prev + (ktBase - IN_DIM));
    const size_t tileH = ((size_t)rb * NKT + kt) * A_TILE_HALVES;
#pragma unroll
    for (int i = 0; i < 2; ++i) {
        int idx = tid + 256 * i;
        int ar  = idx >> 3;
        int c4  = idx & 7;
        const float4 v = *(const float4*)(Asrc + (size_t)(rb * MT + ar) * IN_DIM + c4 * 4);
        unsigned int h01, l01, h23, l23;
        split2(v.x, v.y, h01, l01);
        split2(v.z, v.w, h23, l23);
        *(uint2*)(&wsA[tileH + ar * ASTR + c4 * 4])          = make_uint2(h01, h23);
        *(uint2*)(&wsA[tileH + A_LO_O + ar * ASTR + c4 * 4]) = make_uint2(l01, l23);
    }
}

// =====================================================================
// Prep pass 2: convert gate weights into K-transposed bf16 hi/lo W-tiles
// =====================================================================
__global__ __launch_bounds__(256)
void prep_W(const float* __restrict__ Wf, const float* __restrict__ Wi,
            const float* __restrict__ Wu, const float* __restrict__ Wo,
            unsigned short* __restrict__ wsW) {
    const int tid = threadIdx.x;
    const int kt  = blockIdx.x;
    const int hb  = blockIdx.y;
    const float* Wg[4] = {Wf, Wi, Wu, Wo};
    const int c4 = tid & 15;
    const int ak = (tid >> 4) * 2;
    const size_t tileH = ((size_t)kt * NHB + hb) * W_TILE_HALVES;
#pragma unroll
    for (int g = 0; g < 4; ++g) {
        const float* W = Wg[g] + (size_t)(kt * KT + ak) * H_DIM + hb * NT + c4 * 4;
        const float4 r0 = *(const float4*)(W);
        const float4 r1 = *(const float4*)(W + H_DIM);
        const float r0e[4] = {r0.x, r0.y, r0.z, r0.w};
        const float r1e[4] = {r1.x, r1.y, r1.z, r1.w};
#pragma unroll
        for (int j = 0; j < 4; ++j) {
            unsigned int h, l;
            split2(r0e[j], r1e[j], h, l);
            int col = g * 64 + c4 * 4 + j;
            *(unsigned int*)(&wsW[tileH + col * BSTR + ak])                     = h;
            *(unsigned int*)(&wsW[tileH + (B_LO_O - B_HI_O) + col * BSTR + ak]) = l;
        }
    }
}

// =====================================================================
// A-tile staging: per-lane async copies (ASYNCcnt), 5 x b64 per thread
// =====================================================================
static __device__ __forceinline__ void stage_A_async(unsigned ldsStageByte,
                                                     const unsigned short* aTile,
                                                     int tid) {
    const unsigned voffA = (unsigned)tid * 8u;
    const unsigned ldsA  = ldsStageByte + voffA;
#pragma unroll
    for (int j = 0; j < 5; ++j)
        asm volatile("global_load_async_to_lds_b64 %0, %1, %2 offset:%3"
                     :: "v"(ldsA), "v"(voffA), "s"(aTile), "i"(j * 2048)
                     : "memory");
}

// =====================================================================
// W-tile staging: Tensor Data Mover (TENSORcnt), one instruction/tile.
// 1-D copy: data_size=8B, tile_dim0 = tensor_dim0 = stride = elems8B.
// D# built per ISA ch.8 bit layout; group0.type=2, count=1.
// =====================================================================
static __device__ __forceinline__ void tdm_load(unsigned ldsByte,
                                                const void* gptr,
                                                unsigned elems8B) {
    const unsigned long long ga = (unsigned long long)(size_t)gptr;
    u32x4 g0;
    g0[0] = 0x1u;                                            // count=1, user mode
    g0[1] = ldsByte;                                         // lds_addr
    g0[2] = (unsigned)ga;                                    // global_addr[31:0]
    g0[3] = ((unsigned)(ga >> 32) & 0x01FFFFFFu)             // global_addr[56:32]
            | 0x80000000u;                                   // type=2 ("image")
    u32x8 g1;
    g1[0] = 0x00030000u;                                     // data_size=3 (8B)
    g1[1] = (elems8B & 0xFFFFu) << 16;                       // tensor_dim0[15:0]
    g1[2] = ((elems8B >> 16) & 0xFFFFu) | (1u << 16);        // td0[31:16] | tensor_dim1=1
    g1[3] = (elems8B & 0xFFFFu) << 16;                       // td1 hi=0 | tile_dim0
    g1[4] = 1u;                                              // tile_dim1=1, tile_dim2=0
    g1[5] = elems8B;                                         // tensor_dim0_stride[31:0]
    g1[6] = 0u;                                              // stride hi | td1_stride lo
    g1[7] = 0u;
    asm volatile("tensor_load_to_lds %0, %1" :: "s"(g0), "s"(g1) : "memory");
}

// =====================================================================
// Main fused LSTM kernel: TDM + async double-buffered bf16x3 WMMA GEMM
// grid (NRB, NHB), block 256 (8 waves)
// =====================================================================
__global__ __launch_bounds__(256)
void lstm_cell_wmma_async(const unsigned short* __restrict__ wsA,
                          const unsigned short* __restrict__ wsW,
                          const float* __restrict__ c_prev,
                          const float* __restrict__ bfp, const float* __restrict__ bip,
                          const float* __restrict__ bup, const float* __restrict__ bop,
                          float* __restrict__ out) {
    __shared__ __align__(16) unsigned short lds[2 * STAGE_HALVES];

    const int tid   = threadIdx.x;
    const int lane  = tid & 31;
    const int wave  = tid >> 5;
    const int waveM = wave & 3;
    const int waveN = wave >> 2;
    const int rb = blockIdx.x;
    const int hb = blockIdx.y;

    const unsigned ldsBase = (unsigned)(size_t)(&lds[0]);

    v8f acc[4][2];
#pragma unroll
    for (int g = 0; g < 4; ++g)
#pragma unroll
        for (int nf = 0; nf < 2; ++nf)
            acc[g][nf] = (v8f){0.f, 0.f, 0.f, 0.f, 0.f, 0.f, 0.f, 0.f};

    const unsigned short* aTiles = wsA + (size_t)rb * NKT * A_TILE_HALVES;
    const unsigned short* wTiles = wsW + (size_t)hb * W_TILE_HALVES;

    // prologue: fill buffer 0 with k-tile 0
    stage_A_async(ldsBase, aTiles, tid);
    if (wave == 0)
        tdm_load(ldsBase + (unsigned)A_TILE_BYTES, wTiles, W_TILE_BYTES / 8);

    for (int kti = 0; kti < NKT; ++kti) {
        // my async A copies for the current buffer are done
        asm volatile("s_wait_asynccnt 0x0" ::: "memory");
        // wave 0: the TDM W-tile move is done
        if (wave == 0)
            __builtin_amdgcn_s_wait_tensorcnt(0);
        // publish current buffer; also proves all waves finished the other one
        __syncthreads();
        // overlap: kick off next k-tile into the other buffer
        if (kti + 1 < NKT) {
            const unsigned nb = ldsBase + (unsigned)(((kti + 1) & 1) * STAGE_BYTES);
            stage_A_async(nb, aTiles + (size_t)(kti + 1) * A_TILE_HALVES, tid);
            if (wave == 0)
                tdm_load(nb + (unsigned)A_TILE_BYTES,
                         wTiles + (size_t)(kti + 1) * NHB * W_TILE_HALVES,
                         W_TILE_BYTES / 8);
        }

        const int bufH = (kti & 1) * STAGE_HALVES;
        // ---- WMMA compute on current buffer ----
        const int m  = lane & 15;
        const int kq = (lane >> 4) * 8;
        const int aIdx = bufH + (waveM * 16 + m) * ASTR + kq;
        v16bf aHi = load_frag(&lds[A_HI_O + aIdx], &lds[A_HI_O + aIdx + 16]);
        v16bf aLo = load_frag(&lds[A_LO_O + aIdx], &lds[A_LO_O + aIdx + 16]);

        const int kb = (lane >> 4) * 8;
#pragma unroll
        for (int g = 0; g < 4; ++g) {
#pragma unroll
            for (int nf = 0; nf < 2; ++nf) {
                int col  = g * 64 + waveN * 32 + nf * 16 + m;
                int bIdx = bufH + col * BSTR + kb;
                v16bf bHi = load_frag(&lds[B_HI_O + bIdx], &lds[B_HI_O + bIdx + 16]);
                v16bf bLo = load_frag(&lds[B_LO_O + bIdx], &lds[B_LO_O + bIdx + 16]);
                acc[g][nf] = __builtin_amdgcn_wmma_f32_16x16x32_bf16(
                    false, aHi, false, bHi, (short)0, acc[g][nf], false, false);
                acc[g][nf] = __builtin_amdgcn_wmma_f32_16x16x32_bf16(
                    false, aHi, false, bLo, (short)0, acc[g][nf], false, false);
                acc[g][nf] = __builtin_amdgcn_wmma_f32_16x16x32_bf16(
                    false, aLo, false, bHi, (short)0, acc[g][nf], false, false);
            }
        }
    }

    // ---- fused LSTM epilogue ----
    const int nC   = lane & 15;
    const int mOff = (lane >> 4) * 8;
    const size_t CT_OFF = (size_t)B_ROWS * H_DIM;
#pragma unroll
    for (int nf = 0; nf < 2; ++nf) {
        const int h = hb * NT + waveN * 32 + nf * 16 + nC;
        const float bF = bfp[h], bI = bip[h], bU = bup[h], bO = bop[h];
#pragma unroll
        for (int i = 0; i < 8; ++i) {
            const int row = rb * MT + waveM * 16 + mOff + i;
            const size_t o = (size_t)row * H_DIM + h;
            const float fg = fast_sigmoid(acc[0][nf][i] + bF);
            const float ig = fast_sigmoid(acc[1][nf][i] + bI);
            const float gg = fast_tanh(acc[2][nf][i] + bU);
            const float og = fast_sigmoid(acc[3][nf][i] + bO);
            const float c  = c_prev[o] * fg + ig * gg;
            out[o]          = og * fast_tanh(c);
            out[CT_OFF + o] = c;
        }
    }
}

// =====================================================================
// Fallback (self-contained) for small ws
// =====================================================================
__global__ __launch_bounds__(256)
void lstm_cell_wmma_fallback(const float* __restrict__ x, const float* __restrict__ h_prev,
                             const float* __restrict__ c_prev,
                             const float* __restrict__ Wf, const float* __restrict__ Wi,
                             const float* __restrict__ Wu, const float* __restrict__ Wo,
                             const float* __restrict__ bfp, const float* __restrict__ bip,
                             const float* __restrict__ bup, const float* __restrict__ bop,
                             float* __restrict__ out) {
    __shared__ __align__(16) unsigned short lds[STAGE_HALVES];
    const int tid   = threadIdx.x;
    const int lane  = tid & 31;
    const int wave  = tid >> 5;
    const int waveM = wave & 3;
    const int waveN = wave >> 2;
    const int rowBase = blockIdx.x * MT;
    const int hBase   = blockIdx.y * NT;
    const float* Wg[4] = {Wf, Wi, Wu, Wo};

    v8f acc[4][2];
#pragma unroll
    for (int g = 0; g < 4; ++g)
#pragma unroll
        for (int nf = 0; nf < 2; ++nf)
            acc[g][nf] = (v8f){0.f, 0.f, 0.f, 0.f, 0.f, 0.f, 0.f, 0.f};

    const int bc4 = tid & 15;
    const int bak = (tid >> 4) * 2;

    for (int kt = 0; kt < K_DIM; kt += KT) {
        const float* Asrc = (kt < IN_DIM) ? (x + kt) : (h_prev + (kt - IN_DIM));
#pragma unroll
        for (int i = 0; i < 2; ++i) {
            int idx = tid + 256 * i;
            int ar  = idx >> 3;
            int c4  = idx & 7;
            const float4 v = *(const float4*)(Asrc + (size_t)(rowBase + ar) * IN_DIM + c4 * 4);
            unsigned int h01, l01, h23, l23;
            split2(v.x, v.y, h01, l01);
            split2(v.z, v.w, h23, l23);
            *(uint2*)(&lds[A_HI_O + ar * ASTR + c4 * 4]) = make_uint2(h01, h23);
            *(uint2*)(&lds[A_LO_O + ar * ASTR + c4 * 4]) = make_uint2(l01, l23);
        }
#pragma unroll
        for (int g = 0; g < 4; ++g) {
            const float* W = Wg[g] + (size_t)(kt + bak) * H_DIM + hBase + bc4 * 4;
            const float4 r0 = *(const float4*)(W);
            const float4 r1 = *(const float4*)(W + H_DIM);
            const float r0e[4] = {r0.x, r0.y, r0.z, r0.w};
            const float r1e[4] = {r1.x, r1.y, r1.z, r1.w};
#pragma unroll
            for (int j = 0; j < 4; ++j) {
                unsigned int h, l;
                split2(r0e[j], r1e[j], h, l);
                int col = g * 64 + bc4 * 4 + j;
                *(unsigned int*)(&lds[B_HI_O + col * BSTR + bak]) = h;
                *(unsigned int*)(&lds[B_LO_O + col * BSTR + bak]) = l;
            }
        }
        __syncthreads();
        {
            const int m  = lane & 15;
            const int kq = (lane >> 4) * 8;
            const int aIdx = (waveM * 16 + m) * ASTR + kq;
            v16bf aHi = load_frag(&lds[A_HI_O + aIdx], &lds[A_HI_O + aIdx + 16]);
            v16bf aLo = load_frag(&lds[A_LO_O + aIdx], &lds[A_LO_O + aIdx + 16]);
            const int kb = (lane >> 4) * 8;
#pragma unroll
            for (int g = 0; g < 4; ++g) {
#pragma unroll
                for (int nf = 0; nf < 2; ++nf) {
                    int col  = g * 64 + waveN * 32 + nf * 16 + m;
                    int bIdx = col * BSTR + kb;
                    v16bf bHi = load_frag(&lds[B_HI_O + bIdx], &lds[B_HI_O + bIdx + 16]);
                    v16bf bLo = load_frag(&lds[B_LO_O + bIdx], &lds[B_LO_O + bIdx + 16]);
                    acc[g][nf] = __builtin_amdgcn_wmma_f32_16x16x32_bf16(
                        false, aHi, false, bHi, (short)0, acc[g][nf], false, false);
                    acc[g][nf] = __builtin_amdgcn_wmma_f32_16x16x32_bf16(
                        false, aHi, false, bLo, (short)0, acc[g][nf], false, false);
                    acc[g][nf] = __builtin_amdgcn_wmma_f32_16x16x32_bf16(
                        false, aLo, false, bHi, (short)0, acc[g][nf], false, false);
                }
            }
        }
        __syncthreads();
    }

    const int nC   = lane & 15;
    const int mOff = (lane >> 4) * 8;
    const size_t CT_OFF = (size_t)B_ROWS * H_DIM;
#pragma unroll
    for (int nf = 0; nf < 2; ++nf) {
        const int h = hBase + waveN * 32 + nf * 16 + nC;
        const float bF = bfp[h], bI = bip[h], bU = bup[h], bO = bop[h];
#pragma unroll
        for (int i = 0; i < 8; ++i) {
            const int row = rowBase + waveM * 16 + mOff + i;
            const size_t o = (size_t)row * H_DIM + h;
            const float fg = fast_sigmoid(acc[0][nf][i] + bF);
            const float ig = fast_sigmoid(acc[1][nf][i] + bI);
            const float gg = fast_tanh(acc[2][nf][i] + bU);
            const float og = fast_sigmoid(acc[3][nf][i] + bO);
            const float c  = c_prev[o] * fg + ig * gg;
            out[o]          = og * fast_tanh(c);
            out[CT_OFF + o] = c;
        }
    }
}

extern "C" void kernel_launch(void* const* d_in, const int* in_sizes, int n_in,
                              void* d_out, int out_size, void* d_ws, size_t ws_size,
                              hipStream_t stream) {
    const float* x      = (const float*)d_in[0];
    const float* h_prev = (const float*)d_in[1];
    const float* c_prev = (const float*)d_in[2];
    // d_in[3] = embed (unused by reference forward)
    const float* Wf = (const float*)d_in[4];
    const float* bf = (const float*)d_in[5];
    const float* Wi = (const float*)d_in[6];
    const float* bi = (const float*)d_in[7];
    const float* Wu = (const float*)d_in[8];
    const float* bu = (const float*)d_in[9];
    const float* Wo = (const float*)d_in[10];
    const float* bo = (const float*)d_in[11];
    float* out = (float*)d_out;

    if (ws_size >= WS_NEEDED) {
        unsigned short* wsA = (unsigned short*)d_ws;
        unsigned short* wsW = (unsigned short*)((char*)d_ws + WS_A_BYTES);
        prep_A<<<dim3(NRB, NKT), 256, 0, stream>>>(x, h_prev, wsA);
        prep_W<<<dim3(NKT, NHB), 256, 0, stream>>>(Wf, Wi, Wu, Wo, wsW);
        lstm_cell_wmma_async<<<dim3(NRB, NHB), 256, 0, stream>>>(
            wsA, wsW, c_prev, bf, bi, bu, bo, out);
    } else {
        lstm_cell_wmma_fallback<<<dim3(NRB, NHB), 256, 0, stream>>>(
            x, h_prev, c_prev, Wf, Wi, Wu, Wo, bf, bi, bu, bo, out);
    }
}
